// SimpleLstmModel_66760971649617
// MI455X (gfx1250) — matrix operand order repaired and checked by reading, not compiled, not run on hardware
//
#include <hip/hip_runtime.h>
#include <hip/hip_bf16.h>

// Problem constants
static constexpr int kB  = 256;
static constexpr int kT  = 512;
static constexpr int kI  = 64;
static constexpr int kH  = 256;
static constexpr int kNG = 4 * kH;   // 1024 gate columns (i|f|g|o)

typedef __attribute__((ext_vector_type(16))) __bf16 v16bf;
typedef __attribute__((ext_vector_type(8)))  float  v8f;

#if __has_builtin(__builtin_amdgcn_sched_barrier)
#define SCHED_FENCE() __builtin_amdgcn_sched_barrier(0)
#else
#define SCHED_FENCE()
#endif

union BF16Frag {
    uint4 q[2];
    v16bf v;
};

__device__ inline unsigned short f2bf(float x) {
    unsigned int u = __builtin_bit_cast(unsigned int, x);
    unsigned int r = u + 0x7fffu + ((u >> 16) & 1u);   // round-to-nearest-even
    return (unsigned short)(r >> 16);
}
__device__ inline float bf2f(unsigned short v) {
    unsigned int u = ((unsigned int)v) << 16;
    return __builtin_bit_cast(float, u);
}
__device__ inline float sigm(float x) { return 1.0f / (1.0f + __expf(-x)); }

// A fragment (16x32 bf16, row-major source): lane holds row = lane&15.
// lanes 0-15: K [k0..k0+15]
// lanes 16-31: K [k0+16..k0+31]
__device__ inline v16bf load_a_frag(const unsigned short* rowk0, int laneHalf) {
    const uint4* p = (const uint4*)rowk0;        // 8 bf16 per uint4
    BF16Frag f;
    f.q[0] = p[2 * laneHalf];
    f.q[1] = p[2 * laneHalf + 1];
    return f.v;
}

// B fragment (32x16 bf16): lane holds column n = lane&15, 16 contiguous K
// starting at k0 + (lane>>4)*16. Pointer already offset to that position.
__device__ inline v16bf load_b_frag(const unsigned short* p16) {
    const uint4* p = (const uint4*)p16;
    BF16Frag f;
    f.q[0] = p[0];
    f.q[1] = p[1];
    return f.v;
}

__global__ void cvt_f32_bf16(const float* __restrict__ in,
                             unsigned short* __restrict__ out, int n) {
    int i = blockIdx.x * 256 + threadIdx.x;
    if (i < n) out[i] = f2bf(in[i]);
}

// One persistent workgroup per (16-batch-row tile, direction).
// Block = 256 threads = 8 waves; wave w owns gate columns [w*128, w*128+128).
// DIN is the input width (64 for layer 0, 512 for layer 1) -> full unroll,
// static instruction offsets, pipelined loads.
template <int DIN>
__global__ __launch_bounds__(256)
void lstm_bidir_wmma(const unsigned short* __restrict__ xin,
                     const unsigned short* __restrict__ wih_f,
                     const unsigned short* __restrict__ whh_f,
                     const float* __restrict__ bih_f, const float* __restrict__ bhh_f,
                     const unsigned short* __restrict__ wih_r,
                     const unsigned short* __restrict__ whh_r,
                     const float* __restrict__ bih_r, const float* __restrict__ bhh_r,
                     unsigned short* __restrict__ hout,    // [B,T,2H] bf16 or null
                     unsigned short* __restrict__ hlast,   // [B,2H] bf16 at t=T-1 or null
                     int steps_fwd, int steps_bwd) {
    __shared__ __align__(16) unsigned short gbuf[16][kNG];   // 32 KB gate pre-activations
    __shared__ __align__(16) unsigned short hbuf[16][kH];    // 8 KB hidden state
    __shared__ float biasbuf[kNG];                           // 4 KB

    const int tid    = threadIdx.x;
    const int lane   = tid & 31;
    const int wave   = tid >> 5;
    const int dir    = blockIdx.y;
    const int batch0 = blockIdx.x * 16;

    const unsigned short* wih = dir ? wih_r : wih_f;
    const unsigned short* whh = dir ? whh_r : whh_f;
    const float* bih = dir ? bih_r : bih_f;
    const float* bhh = dir ? bhh_r : bhh_f;
    const int steps = dir ? steps_bwd : steps_fwd;

    for (int i = tid; i < kNG; i += 256) biasbuf[i] = bih[i] + bhh[i];
    for (int i = tid; i < 16 * kH; i += 256) (&hbuf[0][0])[i] = 0;
    float c[16];
#pragma unroll
    for (int i = 0; i < 16; ++i) c[i] = 0.0f;
    __syncthreads();

    const int laneHalf = lane >> 4;       // 0/1
    const int lrow     = lane & 15;       // A row / B column within a tile
    const int n0       = wave * 128;      // wave's gate-column base
    const int em       = tid >> 4;        // elementwise row (0..15)
    const int ej0      = (tid & 15) << 4; // elementwise column base (x16)

    // --- Time-invariant per-lane state ---
    // Single fragment base pointer per weight matrix; per-(nt,k0) offsets are
    // compile-time constants folded into the load instruction's 24-bit imm:
    //   addr(nt, k0) = base + nt*16*K + k0        (elements)
    const unsigned short* wbX = wih + (size_t)(n0 + lrow) * DIN + (laneHalf << 4);
    const unsigned short* wbH = whh + (size_t)(n0 + lrow) * kH  + (laneHalf << 4);
    float bv[8];
#pragma unroll
    for (int nt = 0; nt < 8; ++nt) bv[nt] = biasbuf[n0 + nt * 16 + lrow];
    // Per-lane x base (row = batch0 + lrow); add t*DIN each step.
    const unsigned short* xbase = xin + (size_t)(batch0 + lrow) * kT * DIN;
    const unsigned short* hrow  = &hbuf[lrow][0];

    for (int s = 0; s < steps; ++s) {
        const int t = dir ? (kT - 1 - s) : s;

        // Opaque per-iteration offset (always 0): stops LICM from hoisting the
        // loop-invariant weight loads out of the time loop (which would spill
        // the whole weight slice to scratch), while keeping the pointers
        // derived from kernel args so loads stay global_load (not flat) and
        // only tick LOADcnt.
        unsigned oX = 0, oH = 0;
        asm volatile("" : "+v"(oX), "+v"(oH));
        const unsigned short* wx = wbX + oX;
        const unsigned short* wh = wbH + oH;

        // Accumulators: gates[16, n0..n0+128) initialized with combined bias.
        v8f acc[8];
#pragma unroll
        for (int nt = 0; nt < 8; ++nt) {
#pragma unroll
            for (int v = 0; v < 8; ++v) acc[nt][v] = bv[nt];
        }

        // ---- Input projection: x_t [16, DIN] @ w_ih^T ----
        const unsigned short* xrow = xbase + (size_t)t * DIN;
#pragma unroll
        for (int ks = 0; ks < DIN / 32; ++ks) {
            const int k0 = ks * 32;
            v16bf a = load_a_frag(xrow + k0, laneHalf);
            v16bf b[8];
#pragma unroll
            for (int nt = 0; nt < 8; ++nt)
                b[nt] = load_b_frag(wx + nt * 16 * DIN + k0);
            // All 18 128-bit loads issued before any WMMA of this K-step: forces 8
            // distinct fragment buffers -> staggered s_wait_loadcnt instead of
            // a full wait before every WMMA. Next K-step's loads can still be
            // hoisted above this K-step's WMMAs (they are in the next region).
            SCHED_FENCE();
#pragma unroll
            for (int nt = 0; nt < 8; ++nt)
                acc[nt] = __builtin_amdgcn_wmma_f32_16x16x32_bf16(
                    false, a, false, b[nt], (short)0, acc[nt], false, false);
        }

        // ---- Recurrent projection: h [16, 256] @ w_hh^T (h from LDS) ----
#pragma unroll
        for (int ks = 0; ks < kH / 32; ++ks) {
            const int k0 = ks * 32;
            v16bf a = load_a_frag(hrow + k0, laneHalf);
            v16bf b[8];
#pragma unroll
            for (int nt = 0; nt < 8; ++nt)
                b[nt] = load_b_frag(wh + nt * 16 * kH + k0);
            SCHED_FENCE();
#pragma unroll
            for (int nt = 0; nt < 8; ++nt)
                acc[nt] = __builtin_amdgcn_wmma_f32_16x16x32_bf16(
                    false, a, false, b[nt], (short)0, acc[nt], false, false);
        }

        // Spill gates to LDS (C/D layout: VGPR v -> row v + 8*(lane>=16), col lane%16)
#pragma unroll
        for (int nt = 0; nt < 8; ++nt) {
            int col = n0 + nt * 16 + lrow;
#pragma unroll
            for (int v = 0; v < 8; ++v) {
                int m = v + (laneHalf << 3);
                gbuf[m][col] = f2bf(acc[nt][v]);
            }
        }
        __syncthreads();

        // Elementwise LSTM cell: thread owns row em, cols [ej0, ej0+16)
#pragma unroll
        for (int jj = 0; jj < 16; ++jj) {
            int j = ej0 + jj;
            float iv = sigm(bf2f(gbuf[em][j]));
            float fv = sigm(bf2f(gbuf[em][kH + j]));
            float gv = tanhf(bf2f(gbuf[em][2 * kH + j]));
            float ov = sigm(bf2f(gbuf[em][3 * kH + j]));
            float cn = fv * c[jj] + iv * gv;
            c[jj] = cn;
            float hv = ov * tanhf(cn);
            unsigned short hb = f2bf(hv);
            hbuf[em][j] = hb;
            if (hout)
                hout[((size_t)(batch0 + em) * kT + t) * (2 * kH) + dir * kH + j] = hb;
            if (hlast && t == kT - 1)
                hlast[(size_t)(batch0 + em) * (2 * kH) + dir * kH + j] = hb;
        }
        __syncthreads();
    }
}

// out[b] = sum_j hlast[b,j] * fc_w[j] + fc_b   (O = 1)
__global__ void fc_kernel(const unsigned short* __restrict__ hlast,
                          const float* __restrict__ fcw,
                          const float* __restrict__ fcb,
                          float* __restrict__ out) {
    int b = blockIdx.x * blockDim.x + threadIdx.x;
    if (b >= kB) return;
    float s = fcb[0];
    for (int j = 0; j < 2 * kH; ++j)
        s += bf2f(hlast[(size_t)b * (2 * kH) + j]) * fcw[j];
    out[b] = s;
}

extern "C" void kernel_launch(void* const* d_in, const int* in_sizes, int n_in,
                              void* d_out, int out_size, void* d_ws, size_t ws_size,
                              hipStream_t stream) {
    (void)in_sizes; (void)n_in; (void)out_size; (void)ws_size;

    // Workspace carve-out (bf16 buffers), 256B aligned
    size_t off = 0;
    auto take = [&](size_t elems) {
        unsigned short* p = (unsigned short*)((char*)d_ws + off);
        off += ((elems * 2) + 255) & ~(size_t)255;
        return p;
    };
    unsigned short* xb     = take((size_t)kB * kT * kI);       // 16 MB
    unsigned short* w0f    = take((size_t)kNG * kI);
    unsigned short* w0r    = take((size_t)kNG * kI);
    unsigned short* wh0f   = take((size_t)kNG * kH);
    unsigned short* wh0r   = take((size_t)kNG * kH);
    unsigned short* w1f    = take((size_t)kNG * 2 * kH);
    unsigned short* w1r    = take((size_t)kNG * 2 * kH);
    unsigned short* wh1f   = take((size_t)kNG * kH);
    unsigned short* wh1r   = take((size_t)kNG * kH);
    unsigned short* hcat0  = take((size_t)kB * kT * 2 * kH);   // 128 MB
    unsigned short* hlast1 = take((size_t)kB * 2 * kH);

    auto cvt = [&](int idx, unsigned short* out, size_t n) {
        cvt_f32_bf16<<<(unsigned)((n + 255) / 256), 256, 0, stream>>>(
            (const float*)d_in[idx], out, (int)n);
    };
    cvt(0,  xb,   (size_t)kB * kT * kI);
    cvt(1,  w0f,  (size_t)kNG * kI);
    cvt(2,  wh0f, (size_t)kNG * kH);
    cvt(5,  w0r,  (size_t)kNG * kI);
    cvt(6,  wh0r, (size_t)kNG * kH);
    cvt(9,  w1f,  (size_t)kNG * 2 * kH);
    cvt(10, wh1f, (size_t)kNG * kH);
    cvt(13, w1r,  (size_t)kNG * 2 * kH);
    cvt(14, wh1r, (size_t)kNG * kH);

    dim3 grid(kB / 16, 2), block(256);
    // Layer 0: both directions, full T, write concat output for layer 1.
    lstm_bidir_wmma<kI><<<grid, block, 0, stream>>>(
        xb,
        w0f, wh0f, (const float*)d_in[3],  (const float*)d_in[4],
        w0r, wh0r, (const float*)d_in[7],  (const float*)d_in[8],
        hcat0, nullptr, kT, kT);
    // Layer 1: fwd full T; bwd needs only its t=T-1 output (first reverse step).
    lstm_bidir_wmma<2 * kH><<<grid, block, 0, stream>>>(
        hcat0,
        w1f, wh1f, (const float*)d_in[11], (const float*)d_in[12],
        w1r, wh1r, (const float*)d_in[15], (const float*)d_in[16],
        nullptr, hlast1, kT, 1);

    fc_kernel<<<1, 256, 0, stream>>>(hlast1, (const float*)d_in[17],
                                     (const float*)d_in[18], (float*)d_out);
}
